// CrossRegionAttention_25812753449097
// MI455X (gfx1250) — matrix operand order repaired and checked by reading, not compiled, not run on hardware
//
#include <hip/hip_runtime.h>

typedef unsigned short u16;
typedef __attribute__((ext_vector_type(16))) __bf16 v16bf;
typedef __attribute__((ext_vector_type(8)))  __bf16 v8bf;
typedef __attribute__((ext_vector_type(8)))  float  v8f;

#define DEV static __device__ __forceinline__

// CDNA5 async global->LDS copy path (ASYNCcnt), with portable fallback.
#if __has_builtin(__builtin_amdgcn_global_load_async_to_lds_b128) && \
    __has_builtin(__builtin_amdgcn_s_wait_asynccnt)
#define USE_ASYNC 1
#else
#define USE_ASYNC 0
#endif

// exact operand types for the async builtin: int4 pointers in AS1 (global) / AS3 (LDS)
typedef int i4 __attribute__((vector_size(16)));
typedef __attribute__((address_space(1))) i4* p1_i4;
typedef __attribute__((address_space(3))) i4* p3_i4;

DEV void stage16(const u16* __restrict__ src, u16* __restrict__ dst) {
#if USE_ASYNC
  u16* s = const_cast<u16*>(src);
  __builtin_amdgcn_global_load_async_to_lds_b128((p1_i4)(i4*)(void*)s,
                                                 (p3_i4)(i4*)(void*)dst, 0, 0);
#else
  *(uint4*)dst = *(const uint4*)src;
#endif
}

DEV void async_drain() {
#if USE_ASYNC
  __builtin_amdgcn_s_wait_asynccnt(0);
#endif
}

DEV u16 f2bf(float f) {
  unsigned u = __builtin_bit_cast(unsigned, f);
  unsigned r = u + 0x7fffu + ((u >> 16) & 1u);
  return (u16)(r >> 16);
}

DEV float sigmoidf(float x) { return 1.f / (1.f + __expf(-x)); }

// A fragment 16x32 bf16 at tile origin p (row0,k0 already applied), row stride ld elems.
// Contiguous k order used here (B loaders match): lanes 0-15 row M=lane, K=0..15; lanes 16-31 row M=lane-16, K=16..31
DEV v16bf wmma_load_a(const u16* p, int ld, int lane) {
  int m  = lane & 15;
  int kb = (lane >> 4) << 4;             // 0 or 16
  const u16* r = p + (size_t)m * ld + kb;
  v8bf lo = *(const v8bf*)r;             // K = kb .. kb+7
  v8bf hi = *(const v8bf*)(r + 8);       // K = kb+8 .. kb+15
  v16bf a;
#pragma unroll
  for (int i = 0; i < 8; ++i) { a[i] = lo[i]; a[8 + i] = hi[i]; }
  return a;
}

// B fragment 32x16 where Bop[k][n] = p[n*ld + k]  (operand stored row-major [N][K])
DEV v16bf wmma_load_b_rowT(const u16* p, int ld, int lane) {
  int n  = lane & 15;
  int kb = (lane >> 4) << 4;             // 0 or 16
  return *(const v16bf*)(p + (size_t)n * ld + kb);
}

// B fragment 32x16 where Bop[k][n] = p[k*ld + n]  (operand stored row-major [K][N])
DEV v16bf wmma_load_b_strided(const u16* p, int ld, int lane) {
  int n  = lane & 15;
  int kb = (lane >> 4) << 4;
  const u16* q = p + (size_t)kb * ld + n;
  v16bf b;
#pragma unroll
  for (int j = 0; j < 16; ++j) b[j] = __builtin_bit_cast(__bf16, q[(size_t)j * ld]);
  return b;
}

// ---------------- prep kernels ----------------

__global__ void k_f2bf(const float* __restrict__ src, u16* __restrict__ dst, int n) {
  int i = blockIdx.x * 256 + threadIdx.x;
  if (i < n) dst[i] = f2bf(src[i]);
}

// cw [O, CIN, 3, 3] fp32 -> wr [O][9*CIN] bf16 with k = (dy*3+dx)*CIN + ci
__global__ void k_rearr(const float* __restrict__ src, u16* __restrict__ dst, int cin) {
  int idx = blockIdx.x * 256 + threadIdx.x;
  int ksz = 9 * cin;
  int o = idx / ksz, rem = idx % ksz;
  int t9 = rem / cin, ci = rem % cin;
  int dy = t9 / 3, dx = t9 % 3;
  dst[idx] = f2bf(src[(((size_t)o * cin + ci) * 3 + dy) * 3 + dx]);
}

__global__ void k_bnfold(const float* g1, const float* b1, const float* m1, const float* v1, const float* cb1,
                         const float* g2, const float* b2, const float* m2, const float* v2, const float* cb2,
                         float* sc1, float* sh1, float* sc2, float* sh2) {
  int c = threadIdx.x;
  float s1 = g1[c] * rsqrtf(v1[c] + 1e-5f);
  sc1[c] = s1; sh1[c] = b1[c] + (cb1[c] - m1[c]) * s1;
  float s2 = g2[c] * rsqrtf(v2[c] + 1e-5f);
  sc2[c] = s2; sh2[c] = b2[c] + (cb2[c] - m2[c]) * s2;
}

// x [B,C,64,64] fp32 NCHW -> xu [T=65536, 256] bf16 tokens (t=(b,n,p))
__global__ void k_unfold(const float* __restrict__ x, u16* __restrict__ xu) {
  int idx = blockIdx.x * 256 + threadIdx.x;
  int t = idx >> 8, c = idx & 255;
  int b = t >> 12, n = (t >> 6) & 63, pp = t & 63;
  int w = ((n >> 3) << 3) + (pp >> 3);
  int h = ((n & 7) << 3) + (pp & 7);
  xu[idx] = f2bf(x[(((size_t)b * 256 + c) * 64 + w) * 64 + h]);
}

// ---------------- GEMM: out[t,o] = scale * sum_c A[t,c] * Wt[o,c] ----------------
// M = 65536 (grid.x = 4096), N = 256 (grid.y = 2, 8 waves), K = 256
// A strip (16x256, 8KB) staged once per block into LDS (async on CDNA5).
template <bool OUT_BF16>
__global__ void k_gemm256(const u16* __restrict__ A, const u16* __restrict__ Wt,
                          void* __restrict__ Out, float scale) {
  __shared__ u16 As[16 * 256];
  int tid  = threadIdx.x;
  int lane = tid & 31;
  int wave = tid >> 5;
  int row0 = blockIdx.x * 16;
  int n0   = (blockIdx.y * 8 + wave) * 16;

  const u16* asrc = A + (size_t)row0 * 256;
#pragma unroll
  for (int q = 0; q < 2; ++q) {
    int idx = (tid + q * 256) * 8;       // 512 chunks of 8 u16
    stage16(asrc + idx, &As[idx]);
  }
  async_drain();
  __syncthreads();

  const u16* wrow = Wt + (size_t)(n0 + (lane & 15)) * 256;
  v8f acc = {0.f, 0.f, 0.f, 0.f, 0.f, 0.f, 0.f, 0.f};
#pragma unroll
  for (int k0 = 0; k0 < 256; k0 += 32) {
    if (k0 < 224) __builtin_prefetch(wrow + k0 + 32, 0, 1);
    v16bf a = wmma_load_a(&As[k0], 256, lane);
    v16bf b = wmma_load_b_rowT(Wt + (size_t)n0 * 256 + k0, 256, lane);
    acc = __builtin_amdgcn_wmma_f32_16x16x32_bf16(false, a, false, b, (short)0, acc, false, false);
  }
  int n  = n0 + (lane & 15);
  int mb = row0 + ((lane >> 4) << 3);
#pragma unroll
  for (int r = 0; r < 8; ++r) {
    float v = acc[r] * scale;
    size_t o = (size_t)(mb + r) * 256 + n;
    if (OUT_BF16) ((u16*)Out)[o] = f2bf(v);
    else          ((float*)Out)[o] = v;
  }
}

// ---------------- fused attention: 8192 groups of softmax(Q K^T) V, 64x32 each ----------------
// mode 0 (intra): base g*2048, row stride 32.  mode 1 (cross): base (g>>6)*131072+(g&63)*32, rs 2048.
__global__ void k_attn(const u16* __restrict__ Q, const u16* __restrict__ Km,
                       const u16* __restrict__ Vm, u16* __restrict__ O, int mode) {
  __shared__ float S[64 * 64];
  __shared__ u16   P[64 * 64];
  int g = blockIdx.x;
  int lane = threadIdx.x & 31;
  int wave = threadIdx.x >> 5;   // 4 waves / block
  int strip = wave * 16;
  size_t base; int rs;
  if (mode == 0) { base = (size_t)g * 2048; rs = 32; }
  else           { base = (size_t)(g >> 6) * 131072 + (size_t)(g & 63) * 32; rs = 2048; }
  const u16* Qg = Q + base;
  const u16* Kg = Km + base;
  const u16* Vg = Vm + base;

  // S = Q K^T  (Q pre-scaled); K dim = d = 32 -> one WMMA per 16x16 tile
  v16bf aq = wmma_load_a(Qg + (size_t)strip * rs, rs, lane);
#pragma unroll
  for (int nt = 0; nt < 4; ++nt) {
    v16bf b = wmma_load_b_rowT(Kg + (size_t)(nt * 16) * rs, rs, lane);
    v8f acc = {0.f, 0.f, 0.f, 0.f, 0.f, 0.f, 0.f, 0.f};
    acc = __builtin_amdgcn_wmma_f32_16x16x32_bf16(false, aq, false, b, (short)0, acc, false, false);
    int m0 = strip + ((lane >> 4) << 3);
    int n  = nt * 16 + (lane & 15);
#pragma unroll
    for (int r = 0; r < 8; ++r) S[(m0 + r) * 64 + n] = acc[r];
  }
  __syncthreads();

  // fp32 softmax over rows of S, write bf16 P
  if (threadIdx.x < 64) {
    int row = threadIdx.x;
    float mx = -3.4e38f;
    for (int j = 0; j < 64; ++j) mx = fmaxf(mx, S[row * 64 + j]);
    float s = 0.f;
    for (int j = 0; j < 64; ++j) { float e = __expf(S[row * 64 + j] - mx); S[row * 64 + j] = e; s += e; }
    float inv = 1.f / s;
    for (int j = 0; j < 64; ++j) P[row * 64 + j] = f2bf(S[row * 64 + j] * inv);
  }
  __syncthreads();

  // O = P V  (K dim = 64 -> 2 WMMA k-steps)
#pragma unroll
  for (int nt = 0; nt < 2; ++nt) {
    v8f acc = {0.f, 0.f, 0.f, 0.f, 0.f, 0.f, 0.f, 0.f};
#pragma unroll
    for (int ks = 0; ks < 2; ++ks) {
      v16bf a = wmma_load_a(&P[strip * 64 + ks * 32], 64, lane);
      v16bf b = wmma_load_b_strided(Vg + (size_t)(ks * 32) * rs + nt * 16, rs, lane);
      acc = __builtin_amdgcn_wmma_f32_16x16x32_bf16(false, a, false, b, (short)0, acc, false, false);
    }
    int m0 = strip + ((lane >> 4) << 3);
    int n  = nt * 16 + (lane & 15);
#pragma unroll
    for (int r = 0; r < 8; ++r)
      O[(size_t)g * 2048 + (size_t)(m0 + r) * 32 + n] = f2bf(acc[r]);
  }
}

// ---------------- LayerNorm + sigmoid gate + scatter into NHWC concat buffer ----------------
__global__ void k_ln_inside(const float* __restrict__ proj, const float* __restrict__ bias,
                            const float* __restrict__ gam, const float* __restrict__ bet,
                            const float* __restrict__ x, u16* __restrict__ cat) {
  __shared__ float s1[256], s2[256];
  int t = blockIdx.x, c = threadIdx.x;
  float v = proj[(size_t)t * 256 + c] + bias[c];
  s1[c] = v; s2[c] = v * v;
  __syncthreads();
  for (int off = 128; off > 0; off >>= 1) {
    if (c < off) { s1[c] += s1[c + off]; s2[c] += s2[c + off]; }
    __syncthreads();
  }
  float mean = s1[0] * (1.f / 256.f);
  float var  = s2[0] * (1.f / 256.f) - mean * mean;
  float ln = (v - mean) * rsqrtf(var + 1e-5f) * gam[c] + bet[c];
  int b = t >> 12, n = (t >> 6) & 63, pp = t & 63;
  int w = ((n >> 3) << 3) + (pp >> 3);
  int h = ((n & 7) << 3) + (pp & 7);
  float xv = x[(((size_t)b * 256 + c) * 64 + w) * 64 + h];
  cat[((size_t)(b * 64 + w) * 64 + h) * 512 + c] = f2bf(xv * sigmoidf(ln));
}

__global__ void k_ln_region(const float* __restrict__ proj, const float* __restrict__ bias,
                            const float* __restrict__ gam, const float* __restrict__ bet,
                            const float* __restrict__ x, u16* __restrict__ cat) {
  __shared__ float s1[256], s2[256];
  int t = blockIdx.x, c = threadIdx.x;
  float v = proj[(size_t)t * 256 + c] + bias[c];
  s1[c] = v; s2[c] = v * v;
  __syncthreads();
  for (int off = 128; off > 0; off >>= 1) {
    if (c < off) { s1[c] += s1[c + off]; s2[c] += s2[c + off]; }
    __syncthreads();
  }
  float mean = s1[0] * (1.f / 256.f);
  float var  = s2[0] * (1.f / 256.f) - mean * mean;
  float ln = (v - mean) * rsqrtf(var + 1e-5f) * gam[c] + bet[c];
  int b = t >> 12, pp = (t >> 6) & 63, n = t & 63;
  // replicate the region fold scrambling: j = pp*C + c -> (c_out, kx, ky)
  int b2 = c >> 6, rem = c & 63;
  int kx = rem >> 3, ky = rem & 7;
  int co = pp * 4 + b2;
  int w = ((n >> 3) << 3) + kx;
  int h = ((n & 7) << 3) + ky;
  float xv = x[(((size_t)b * 256 + co) * 64 + w) * 64 + h];
  cat[((size_t)(b * 64 + w) * 64 + h) * 512 + 256 + co] = f2bf(xv * sigmoidf(ln));
}

// ---------------- implicit-GEMM 3x3 SAME conv + BN + ReLU ----------------
// In: NHWC bf16 [16,64,64,CIN]; Wr: [256][9*CIN] bf16; out: bf16 NHWC or fp32 NCHW.
// Input halo tile (3 x 18 x CIN, zero-padded) staged into LDS once per block; all 8
// waves read A fragments from LDS (branch-free inner loop, EXEC all-ones for WMMA).
template <int CIN, bool FINAL>
__global__ void k_conv3(const u16* __restrict__ In, const u16* __restrict__ Wr,
                        const float* __restrict__ scale, const float* __restrict__ shift,
                        void* __restrict__ Out) {
  __shared__ u16 tile[3 * 18 * CIN];     // conv1: 54 KiB, conv2: 27 KiB
  int tid  = threadIdx.x;
  int lane = tid & 31;
  int wave = tid >> 5;
  int mb = blockIdx.x * 16;              // 16 consecutive pixels along h
  int b = mb >> 12, ww = (mb >> 6) & 63, h0 = mb & 63;
  int n0 = (blockIdx.y * 8 + wave) * 16;

  // ---- stage halo tile: iw in [ww-1,ww+1], ih in [h0-1,h0+16], zero-padded ----
  const int CH8 = CIN / 8;               // b128 chunks per (iw,ih) row
  for (int q = tid; q < 54 * CH8; q += 256) {
    int rowi = q / CH8, c8 = q % CH8;
    int iwi = rowi / 18, ihi = rowi % 18;
    int iw = ww + iwi - 1, ih = h0 + ihi - 1;
    u16* dst = &tile[(size_t)rowi * CIN + c8 * 8];
    if (((unsigned)iw < 64u) && ((unsigned)ih < 64u)) {
      stage16(In + ((size_t)(b * 64 + iw) * 64 + ih) * CIN + c8 * 8, dst);
    } else {
      uint4 z = {0u, 0u, 0u, 0u};
      *(uint4*)dst = z;
    }
  }
  async_drain();
  __syncthreads();

  int mlane = lane & 15;
  int kb8 = (lane >> 4) << 4;
  const u16* wrow = Wr + (size_t)(n0 + mlane) * (9 * CIN) + ((lane >> 4) << 4);
  v8f acc = {0.f, 0.f, 0.f, 0.f, 0.f, 0.f, 0.f, 0.f};
#pragma unroll
  for (int dy = 0; dy < 3; ++dy) {
#pragma unroll
    for (int dx = 0; dx < 3; ++dx) {
      const u16* arow = &tile[(size_t)(dy * 18 + (mlane + dx)) * CIN + kb8];
      const u16* wk = wrow + (dy * 3 + dx) * CIN;
      for (int c0 = 0; c0 < CIN; c0 += 32) {
        v8bf lo = *(const v8bf*)(arow + c0);
        v8bf hi = *(const v8bf*)(arow + c0 + 8);
        v16bf a;
#pragma unroll
        for (int i = 0; i < 8; ++i) { a[i] = lo[i]; a[8 + i] = hi[i]; }
        v16bf wb = *(const v16bf*)(wk + c0);
        acc = __builtin_amdgcn_wmma_f32_16x16x32_bf16(false, a, false, wb, (short)0, acc, false, false);
      }
    }
  }
  int n = n0 + mlane;
  float sc = scale[n], sh = shift[n];
  int m0 = (lane >> 4) << 3;
#pragma unroll
  for (int r = 0; r < 8; ++r) {
    int h = h0 + m0 + r;
    float v = fmaxf(acc[r] * sc + sh, 0.f);
    if (FINAL) ((float*)Out)[(((size_t)b * 256 + n) * 64 + ww) * 64 + h] = v;
    else       ((u16*)Out)[((size_t)(b * 64 + ww) * 64 + h) * 256 + n] = f2bf(v);
  }
}

// ---------------- host launch ----------------

extern "C" void kernel_launch(void* const* d_in, const int* in_sizes, int n_in,
                              void* d_out, int out_size, void* d_ws, size_t ws_size,
                              hipStream_t stream) {
  const float* x    = (const float*)d_in[0];
  const float* Wq   = (const float*)d_in[1];
  const float* Wk   = (const float*)d_in[2];
  const float* Wv   = (const float*)d_in[3];
  const float* Wpi  = (const float*)d_in[4];
  const float* bpi  = (const float*)d_in[5];
  const float* lni_g = (const float*)d_in[6];
  const float* lni_b = (const float*)d_in[7];
  const float* Wpr  = (const float*)d_in[8];
  const float* bpr  = (const float*)d_in[9];
  const float* lnr_g = (const float*)d_in[10];
  const float* lnr_b = (const float*)d_in[11];
  const float* cw1  = (const float*)d_in[12];
  const float* cb1  = (const float*)d_in[13];
  const float* bn1_g = (const float*)d_in[14];
  const float* bn1_b = (const float*)d_in[15];
  const float* bn1_m = (const float*)d_in[16];
  const float* bn1_v = (const float*)d_in[17];
  const float* cw2  = (const float*)d_in[18];
  const float* cb2  = (const float*)d_in[19];
  const float* bn2_g = (const float*)d_in[20];
  const float* bn2_b = (const float*)d_in[21];
  const float* bn2_m = (const float*)d_in[22];
  const float* bn2_v = (const float*)d_in[23];

  const size_t Mi = 1048576;
  char* ws = (char*)d_ws;
  u16* xu_h  = (u16*)(ws + 0 * Mi);      // 32 MiB  [65536,256] bf16 tokens
  u16* q_h   = (u16*)(ws + 32 * Mi);     // 32 MiB
  u16* k_h   = (u16*)(ws + 64 * Mi);     // 32 MiB
  u16* v_h   = (u16*)(ws + 96 * Mi);     // 32 MiB
  u16* oi_h  = (u16*)(ws + 128 * Mi);    // 32 MiB  intra attention out
  u16* orr_h = (u16*)(ws + 160 * Mi);    // 32 MiB  cross attention out
  float* proj_f = (float*)(ws + 192 * Mi); // 64 MiB fp32 (reused for both projections)
  u16* cat_h = (u16*)(ws + 256 * Mi);    // 32 MiB  NHWC [16,64,64,512]
  u16* y1_h  = (u16*)(ws + 288 * Mi);    // 16 MiB  NHWC [16,64,64,256]
  char* wbase = ws + 304 * Mi;
  u16* wq_h  = (u16*)(wbase + 0);
  u16* wk_h  = (u16*)(wbase + 131072);
  u16* wv_h  = (u16*)(wbase + 262144);
  u16* wpi_h = (u16*)(wbase + 393216);
  u16* wpr_h = (u16*)(wbase + 524288);
  u16* w1r   = (u16*)(wbase + 655360);   // 2.25 MiB [256][4608]
  u16* w2r   = (u16*)(wbase + 3014656);  // 1.125 MiB [256][2304]
  float* sc1 = (float*)(wbase + 4194304);
  float* sh1 = sc1 + 256;
  float* sc2 = sc1 + 512;
  float* sh2 = sc1 + 768;

  // weight prep
  k_f2bf<<<256, 256, 0, stream>>>(Wq, wq_h, 65536);
  k_f2bf<<<256, 256, 0, stream>>>(Wk, wk_h, 65536);
  k_f2bf<<<256, 256, 0, stream>>>(Wv, wv_h, 65536);
  k_f2bf<<<256, 256, 0, stream>>>(Wpi, wpi_h, 65536);
  k_f2bf<<<256, 256, 0, stream>>>(Wpr, wpr_h, 65536);
  k_rearr<<<4608, 256, 0, stream>>>(cw1, w1r, 512);
  k_rearr<<<2304, 256, 0, stream>>>(cw2, w2r, 256);
  k_bnfold<<<1, 256, 0, stream>>>(bn1_g, bn1_b, bn1_m, bn1_v, cb1,
                                  bn2_g, bn2_b, bn2_m, bn2_v, cb2,
                                  sc1, sh1, sc2, sh2);

  // unfold + QKV projections (q pre-scaled by d^-0.5)
  k_unfold<<<65536, 256, 0, stream>>>(x, xu_h);
  dim3 ggrid(4096, 2);
  k_gemm256<true><<<ggrid, 256, 0, stream>>>(xu_h, wq_h, q_h, 0.17677669529663687f);
  k_gemm256<true><<<ggrid, 256, 0, stream>>>(xu_h, wk_h, k_h, 1.0f);
  k_gemm256<true><<<ggrid, 256, 0, stream>>>(xu_h, wv_h, v_h, 1.0f);

  // intra-patch and cross-region attention (8192 groups each)
  k_attn<<<8192, 128, 0, stream>>>(q_h, k_h, v_h, oi_h, 0);
  k_attn<<<8192, 128, 0, stream>>>(q_h, k_h, v_h, orr_h, 1);

  // inside branch: project, LN, gate, scatter to cat[..., 0:256]
  k_gemm256<false><<<ggrid, 256, 0, stream>>>(oi_h, wpi_h, proj_f, 1.0f);
  k_ln_inside<<<65536, 256, 0, stream>>>(proj_f, bpi, lni_g, lni_b, x, cat_h);

  // region branch: project, LN, gate, scatter to cat[..., 256:512]
  k_gemm256<false><<<ggrid, 256, 0, stream>>>(orr_h, wpr_h, proj_f, 1.0f);
  k_ln_region<<<65536, 256, 0, stream>>>(proj_f, bpr, lnr_g, lnr_b, x, cat_h);

  // conv1 (512->256) + BN + ReLU -> NHWC bf16 ; conv2 (256->256) + BN + ReLU -> NCHW fp32
  k_conv3<512, false><<<ggrid, 256, 0, stream>>>(cat_h, w1r, sc1, sh1, y1_h);
  k_conv3<256, true><<<ggrid, 256, 0, stream>>>(y1_h, w2r, sc2, sh2, d_out);
}
